// VectorQuantizer_5377299055037
// MI455X (gfx1250) — matrix operand order repaired and checked by reading, not compiled, not run on hardware
//
#include <hip/hip_runtime.h>

typedef __attribute__((ext_vector_type(16))) __bf16 v16bf;
typedef __attribute__((ext_vector_type(8)))  float  v8f;
typedef __attribute__((ext_vector_type(4)))  int    v4i;

#define N_ROWS 65536
#define DDIM   256
#define KCOLS  4096

// float -> bf16 bits, round-to-nearest-even
__device__ __forceinline__ unsigned short f2bf(float f) {
  unsigned int u = __float_as_uint(f);
  u += 0x7FFFu + ((u >> 16) & 1u);
  return (unsigned short)(u >> 16);
}

// 16-byte global->LDS copy: async (ASYNCcnt-tracked).
// Builtin signature: param0 = v4i* in AS1 (global), param1 = v4i* in AS3 (LDS),
// then imm offset and cpol.
__device__ __forceinline__ void copy16_g2l(void* lds_p, const void* g_p) {
#if __has_builtin(__builtin_amdgcn_global_load_async_to_lds_b128)
  __builtin_amdgcn_global_load_async_to_lds_b128(
      (__attribute__((address_space(1))) v4i*)(v4i*)const_cast<void*>(g_p),
      (__attribute__((address_space(3))) v4i*)(v4i*)lds_p,
      /*offset=*/0, /*cpol=*/0);
#else
  *(uint4*)lds_p = *(const uint4*)g_p;
#endif
}

__device__ __forceinline__ void wait_async_done() {
#if __has_builtin(__builtin_amdgcn_s_wait_asynccnt)
  __builtin_amdgcn_s_wait_asynccnt((short)0);
#else
  asm volatile("s_wait_asynccnt 0" ::: "memory");
#endif
}

// ---------------------------------------------------------------------------
// Prep: transpose codebook [D,K] f32 -> Bt [K,D] bf16 and c2[k] = sum_d C^2.
// Also zero the loss accumulator slot in d_out.
// ---------------------------------------------------------------------------
__global__ __launch_bounds__(256) void vq_prep(const float* __restrict__ codebook,
                                               unsigned short* __restrict__ Bt,
                                               float* __restrict__ c2,
                                               float* __restrict__ loss_slot) {
  const int c = blockIdx.x * 256 + threadIdx.x;
  if (c == 0) *loss_slot = 0.0f;
  if (c >= KCOLS) return;
  float s = 0.0f;
  for (int d = 0; d < DDIM; ++d) {
    const float v = codebook[(size_t)d * KCOLS + c];
    s += v * v;
    Bt[(size_t)c * DDIM + d] = f2bf(v);
  }
  c2[c] = s;
}

// ---------------------------------------------------------------------------
// Fused GEMM + argmin.
// Block = 256 threads = 8 waves; block M-tile = 256 rows (wave owns 2 M-tiles
// of 16 rows -> two independent WMMA accumulator chains sharing each B frag).
// K processed in chunks of 32 columns; each chunk's B data (16 KB) staged once
// per block into LDS (fragment order) via async global->LDS, double-buffered.
// B fragments are software-pipelined 2 deep so ds_load latency overlaps WMMA.
// score(n,k) = c2[k] - 2*(L@C)[n,k]   (||l||^2 is row-constant)
// ---------------------------------------------------------------------------
__global__ __launch_bounds__(256) void vq_argmin(const float* __restrict__ latents,
                                                 const unsigned short* __restrict__ Bt,
                                                 const float* __restrict__ c2,
                                                 int* __restrict__ ind_i,
                                                 float* __restrict__ ind_f) {
  // Fragment-ordered staging buffers: [buf][ (kt*8 + j)*32 + lane ][16 bf16]
  __shared__ unsigned short Bs[2][32 * DDIM];   // 2 x 16 KB
  __shared__ float c2s[KCOLS];                  // 16 KB

  const int tid = threadIdx.x;
  for (int i = tid; i < KCOLS; i += 256) c2s[i] = c2[i];

  const int wave = tid >> 5;
  const int lane = tid & 31;
  const int half = (lane >> 4) & 1;   // 0: lanes 0-15, 1: lanes 16-31
  const int l15  = lane & 15;
  const int m0   = blockIdx.x * 256 + wave * 32;

  // ---- Preload A fragments for two 16-row tiles (rows m0.., m0+16..).
  // ISA A layout (16-bit, 16x32): lanes 0-15 hold M=lane, K={db..db+7, db+16..23};
  // lanes 16-31 hold M=lane-16, K={db+8..15, db+24..31}.
  union AV { v16bf v; unsigned short u[16]; };
  AV a0[8], a1[8];
#pragma unroll
  for (int t = 0; t < 2; ++t) {
    AV* a = t ? a1 : a0;
    const float* arow = latents + (size_t)(m0 + t * 16 + l15) * DDIM;
#pragma unroll
    for (int j = 0; j < 8; ++j) {
      const int db = j * 32;
      const int k0 = db + (half ? 8 : 0);
      const int k1 = db + 16 + (half ? 8 : 0);
      const float4 f0 = *reinterpret_cast<const float4*>(arow + k0);
      const float4 f1 = *reinterpret_cast<const float4*>(arow + k0 + 4);
      const float4 f2 = *reinterpret_cast<const float4*>(arow + k1);
      const float4 f3 = *reinterpret_cast<const float4*>(arow + k1 + 4);
      a[j].u[0]  = f2bf(f0.x); a[j].u[1]  = f2bf(f0.y);
      a[j].u[2]  = f2bf(f0.z); a[j].u[3]  = f2bf(f0.w);
      a[j].u[4]  = f2bf(f1.x); a[j].u[5]  = f2bf(f1.y);
      a[j].u[6]  = f2bf(f1.z); a[j].u[7]  = f2bf(f1.w);
      a[j].u[8]  = f2bf(f2.x); a[j].u[9]  = f2bf(f2.y);
      a[j].u[10] = f2bf(f2.z); a[j].u[11] = f2bf(f2.w);
      a[j].u[12] = f2bf(f3.x); a[j].u[13] = f2bf(f3.y);
      a[j].u[14] = f2bf(f3.z); a[j].u[15] = f2bf(f3.w);
    }
  }

  // ---- Staging addresses hoisted out of the chunk loop (affine in chunk).
  // Piece p (0..1023, 16 B each): lc = p>>1 selects (kt, j, lane), hi = p&1.
  const unsigned short* gsrc[4];
  unsigned int loff[4];
#pragma unroll
  for (int i = 0; i < 4; ++i) {
    const int p    = tid + 256 * i;
    const int hi   = p & 1;
    const int lc   = p >> 1;
    const int ln   = lc & 31;
    const int j    = (lc >> 5) & 7;
    const int kt   = lc >> 8;
    const int sl15 = ln & 15;
    const int shlf = (ln >> 4) & 1;
    gsrc[i] = Bt + (size_t)(kt * 16 + sl15) * DDIM + j * 32 + hi * 16 + shlf * 8;
    loff[i] = (unsigned)(lc * 16 + hi * 8);
  }
  auto stage_chunk = [&](int chunk, int buf) {
    const size_t coff = (size_t)chunk * 32 * DDIM;   // 16 KB per chunk
#pragma unroll
    for (int i = 0; i < 4; ++i)
      copy16_g2l(&Bs[buf][loff[i]], gsrc[i] + coff);
  };

  stage_chunk(0, 0);
  wait_async_done();
  __syncthreads();

  // Running per-lane argmin; C-tile VGPR v holds row (tilebase + half*8 + v).
  float minv0[8], minv1[8];
  int   mini0[8], mini1[8];
#pragma unroll
  for (int v = 0; v < 8; ++v) {
    minv0[v] = 3.4e38f; minv1[v] = 3.4e38f;
    mini0[v] = 0;       mini1[v] = 0;
  }

  int buf = 0;
  for (int chunk = 0; chunk < KCOLS / 32; ++chunk) {
    if (chunk + 1 < KCOLS / 32) stage_chunk(chunk + 1, buf ^ 1);

    const unsigned short* bufp = &Bs[buf][0];
#pragma unroll
    for (int ktl = 0; ktl < 2; ++ktl) {
      v8f acc0 = {};
      v8f acc1 = {};
      // B layout (16-bit, 32x16): lanes 0-15 hold N=lane, K={0..7, 16..23};
      // lanes 16-31 hold N=lane-16, K={8..15, 24..31}.  Fragment-ordered LDS: per lane
      // 32 contiguous bytes -> two conflict-free ds_load_b128.
      union BV { v16bf v; uint4 q[2]; };
      BV b[2];
      {
        const unsigned short* fb = bufp + (size_t)((ktl * 8) * 32 + lane) * 16;
        b[0].q[0] = *reinterpret_cast<const uint4*>(fb);
        b[0].q[1] = *reinterpret_cast<const uint4*>(fb + 8);
      }
#pragma unroll
      for (int j = 0; j < 8; ++j) {
        if (j < 7) {  // prefetch next fragment into the other buffer slot
          const unsigned short* fb =
              bufp + (size_t)((ktl * 8 + j + 1) * 32 + lane) * 16;
          b[(j + 1) & 1].q[0] = *reinterpret_cast<const uint4*>(fb);
          b[(j + 1) & 1].q[1] = *reinterpret_cast<const uint4*>(fb + 8);
        }
        acc0 = __builtin_amdgcn_wmma_f32_16x16x32_bf16(false, a0[j].v, false,
                                                       b[j & 1].v, (short)0,
                                                       acc0, false, false);
        acc1 = __builtin_amdgcn_wmma_f32_16x16x32_bf16(false, a1[j].v, false,
                                                       b[j & 1].v, (short)0,
                                                       acc1, false, false);
      }
      const int col  = chunk * 32 + ktl * 16 + l15;
      const float cv = c2s[col];
#pragma unroll
      for (int v = 0; v < 8; ++v) {
        const float d0 = cv - 2.0f * acc0[v];
        const float d1 = cv - 2.0f * acc1[v];
        if (d0 < minv0[v]) { minv0[v] = d0; mini0[v] = col; }
        if (d1 < minv1[v]) { minv1[v] = d1; mini1[v] = col; }
      }
    }

    wait_async_done();
    __syncthreads();
    buf ^= 1;
  }

  // Cross-lane reduce within each 16-lane half (masks < 16 stay in-half).
#pragma unroll
  for (int t = 0; t < 2; ++t) {
    float* minv = t ? minv1 : minv0;
    int*   mini = t ? mini1 : mini0;
#pragma unroll
    for (int v = 0; v < 8; ++v) {
      float mv = minv[v];
      int   mi = mini[v];
#pragma unroll
      for (int m = 8; m >= 1; m >>= 1) {
        const float ov = __shfl_xor(mv, m, 32);
        const int   oi = __shfl_xor(mi, m, 32);
        if (ov < mv || (ov == mv && oi < mi)) { mv = ov; mi = oi; }
      }
      if (l15 == 0) {
        const int r = m0 + t * 16 + half * 8 + v;
        ind_i[r] = mi;
        ind_f[r] = (float)mi;
      }
    }
  }
}

// ---------------------------------------------------------------------------
// Gather quantized vectors + fused commitment loss (codebook is L2-resident).
// ---------------------------------------------------------------------------
__global__ __launch_bounds__(256) void vq_gather(const float* __restrict__ latents,
                                                 const float* __restrict__ codebook,
                                                 const int* __restrict__ ind,
                                                 float* __restrict__ quantize,
                                                 float* __restrict__ loss_slot) {
  const size_t i = (size_t)blockIdx.x * 256 + threadIdx.x;
  const int n = (int)(i >> 8);
  const int d = (int)(i & 255);
  const int idx = ind[n];
  const float q = codebook[(size_t)d * KCOLS + idx];
  quantize[i] = q;
  const float df = q - latents[i];
  float p = df * df * (1.0f / ((float)N_ROWS * (float)DDIM));
#pragma unroll
  for (int m = 16; m >= 1; m >>= 1) p += __shfl_xor(p, m, 32);
  if ((threadIdx.x & 31) == 0) atomicAdd(loss_slot, p);
}

// ---------------------------------------------------------------------------
// d_out layout: quantize [N*D] | vq_loss [1] | codebook_ind-as-float [N]
// ws layout:    Bt bf16 [K*D]  | c2 f32 [K]  | ind i32 [N]   (~2.4 MB)
// ---------------------------------------------------------------------------
extern "C" void kernel_launch(void* const* d_in, const int* in_sizes, int n_in,
                              void* d_out, int out_size, void* d_ws, size_t ws_size,
                              hipStream_t stream) {
  const float* latents  = (const float*)d_in[0];
  const float* codebook = (const float*)d_in[1];

  float* out       = (float*)d_out;
  float* quantize  = out;
  float* loss_slot = out + (size_t)N_ROWS * DDIM;
  float* ind_f     = loss_slot + 1;

  unsigned short* Bt = (unsigned short*)d_ws;
  float* c2          = (float*)(Bt + (size_t)KCOLS * DDIM);
  int*   ind_i       = (int*)(c2 + KCOLS);

  vq_prep<<<KCOLS / 256, 256, 0, stream>>>(codebook, Bt, c2, loss_slot);
  vq_argmin<<<N_ROWS / 256, 256, 0, stream>>>(latents, Bt, c2, ind_i, ind_f);
  vq_gather<<<(N_ROWS * DDIM) / 256, 256, 0, stream>>>(latents, codebook, ind_i,
                                                       quantize, loss_slot);
}
